// last_block_65171833749751
// MI455X (gfx1250) — matrix operand order repaired and checked by reading, not compiled, not run on hardware
//
#include <hip/hip_runtime.h>
#include <hip/hip_bf16.h>

// ---------------------------------------------------------------------------
// Deformable bottleneck block on MI455X (gfx1250, wave32).
// All convs are implicit-im2col GEMMs on v_wmma_f32_16x16x32_bf16
// (bf16 inputs, f32 accumulate). Weights pre-repacked to bf16, K-order
// (kh,kw)-major / ci-minor. Main loop is software-pipelined with
// double-buffered LDS: global fetch of tile i+1 overlaps WMMA of tile i,
// one workgroup barrier per K-step. Activation f32->bf16 is one v_add +
// one v_perm_b32 per element pair.
//
// Shapes (fixed by reference): B=4, H=W=64, N = B*H*W = 16384.
// ---------------------------------------------------------------------------

typedef __attribute__((ext_vector_type(16))) __bf16          bf16x16;
typedef __attribute__((ext_vector_type(8)))  float           floatx8;
typedef __attribute__((ext_vector_type(8)))  unsigned short  ushort8;
typedef __attribute__((ext_vector_type(4)))  unsigned int    uint32x4;

union FragU { ushort8 h[2]; bf16x16 v; };

__device__ __forceinline__ unsigned short f32_to_bf16_rne(float f) {
    unsigned int u = __float_as_uint(f);
    u += 0x7FFFu + ((u >> 16) & 1u);     // true round-to-nearest-even (repack path)
    return (unsigned short)(u >> 16);
}

// pack two f32 -> two bf16 in one dword: round-half-up + v_perm_b32
__device__ __forceinline__ unsigned int pack2_bf16(float lo, float hi) {
    unsigned int r0 = __float_as_uint(lo) + 0x8000u;
    unsigned int r1 = __float_as_uint(hi) + 0x8000u;
    // result bytes: [r1.b3, r1.b2, r0.b3, r0.b2]  (src0=r1 -> idx 4..7, src1=r0 -> idx 0..3)
    return __builtin_amdgcn_perm(r1, r0, 0x07060302u);
}

#define RS   40            // LDS row stride in u16 (32 payload + 8 pad; 16B aligned rows)
#define BUFE (128 * RS)    // one tile buffer, in u16 elements (10240 B)

// ---------------------------------------------------------------------------
// Weight repack: f32 OIHW  ->  bf16 [co][khw][ci]   (khw = kh*3+kw)
// ---------------------------------------------------------------------------
__global__ __launch_bounds__(256)
void repack_w(const float* __restrict__ src, unsigned short* __restrict__ dst,
              int Ci, int KHW_) {
    const int idx  = blockIdx.x * 256 + threadIdx.x;     // over Co*Ci*KHW
    const int Kdim = Ci * KHW_;
    const int co   = idx / Kdim;
    const int r    = idx - co * Kdim;
    const int khw  = r / Ci;
    const int ci   = r - khw * Ci;
    dst[idx] = f32_to_bf16_rne(src[(size_t)(co * Ci + ci) * KHW_ + khw]);
}

// KHW: 1 (1x1 conv, pad 0) or 9 (3x3 conv, pad 1)
// EPI: 0 = +bias(p0);  1 = BN(p0=g,p1=b,p2=m,p3=v);  2 = BN + ReLU
template<int KHW, int EPI>
__global__ __launch_bounds__(256)
void conv_gemm_wmma(const unsigned short* __restrict__ Wt,  // bf16 [co][khw][ci]
                    const float* __restrict__ X,
                    float* __restrict__ Out, int Ci, int Kdim, int Co,
                    const float* __restrict__ p0, const float* __restrict__ p1,
                    const float* __restrict__ p2, const float* __restrict__ p3) {
    constexpr int PAD = (KHW == 9) ? 1 : 0;
    __shared__ unsigned short lAs[2 * BUFE];   // A tiles: [m][k] (weights, bf16)
    __shared__ unsigned short lBs[2 * BUFE];   // B tiles: [n][k] (im2col, transposed)

    const int n0 = blockIdx.x * 128;           // spatial (b,y,x) tile base
    const int m0 = blockIdx.y * 128;           // output-channel tile base

    const int t     = threadIdx.x;
    const int lane  = t & 31;
    const int wave  = t >> 5;
    const int waveM = wave >> 1;               // 0..3 -> 32-row slab
    const int waveN = wave & 1;                // 0..1 -> 64-col slab
    const int lm    = lane & 15;
    const int lh    = lane >> 4;

    // staging coordinates (each thread fills 16 contiguous k's of one row)
    const int sRow = t >> 1;                   // 0..127
    const int sK   = (t & 1) * 16;             // 0 or 16
    const int p    = n0 + sRow;                // spatial index for B staging
    const int pb   = p >> 12;
    const int py   = (p >> 6) & 63;
    const int px   = p & 63;

    const floatx8 zf = {0.f, 0.f, 0.f, 0.f, 0.f, 0.f, 0.f, 0.f};
    floatx8 acc[2][4];
#pragma unroll
    for (int i = 0; i < 2; ++i)
#pragma unroll
        for (int j = 0; j < 4; ++j) acc[i][j] = zf;

    // ---- in-flight fetch state (tile being staged) ----
    ushort8 ra0, ra1;       // A: 16 bf16
    float   rb[16];         // B: 16 f32 (converted at commit time)
    bool    rbin = false;

    // issue global loads for K-tile kt_ (no LDS traffic)
    auto fetch = [&](int kt_, int ci0_, int kh_, int kw_) {
        const unsigned short* srcA = Wt + (size_t)(m0 + sRow) * Kdim + kt_ + sK;
        ra0 = *(const ushort8*)(srcA);
        ra1 = *(const ushort8*)(srcA + 8);
        const int iy = py + kh_ - PAD;
        const int ix = px + kw_ - PAD;
        rbin = (KHW == 1) || ((unsigned)iy < 64u && (unsigned)ix < 64u);
        if (rbin) {
            const float* srcB = X + (((size_t)pb * Ci + (ci0_ + sK)) << 12) + (iy << 6) + ix;
#pragma unroll
            for (int j = 0; j < 16; ++j) rb[j] = srcB[(size_t)j << 12];  // imm offs j*16KB
        }
    };

    // convert/pack and store fetched tile into LDS buffer `buf`
    auto commit = [&](int buf) {
        ushort8* dA = (ushort8*)&lAs[buf * BUFE + sRow * RS + sK];
        dA[0] = ra0;
        dA[1] = ra1;
        uint32x4* dB = (uint32x4*)&lBs[buf * BUFE + sRow * RS + sK];
        if (rbin) {
            uint32x4 q0, q1;
#pragma unroll
            for (int j = 0; j < 4; ++j) q0[j] = pack2_bf16(rb[2 * j],     rb[2 * j + 1]);
#pragma unroll
            for (int j = 0; j < 4; ++j) q1[j] = pack2_bf16(rb[2 * j + 8], rb[2 * j + 9]);
            dB[0] = q0;
            dB[1] = q1;
        } else {
            const uint32x4 z = {0u, 0u, 0u, 0u};
            dB[0] = z;
            dB[1] = z;
        }
    };

    // ---- prologue: stage tile 0 into buffer 0 ----
    fetch(0, 0, 0, 0);
    commit(0);

    int ci0 = 0, kh = 0, kw = 0;               // scalar K-tile decomposition
    int cur = 0;
    for (int kt = 0; kt < Kdim; kt += 32) {
        // decomposition of next tile
        int nci0 = ci0 + 32, nkh = kh, nkw = kw;
        if (nci0 == Ci) { nci0 = 0; if (KHW == 9) { if (++nkw == 3) { nkw = 0; ++nkh; } } }

        __syncthreads();                       // buf[cur] stores visible; buf[cur^1] readers done

        const bool more = (kt + 32) < Kdim;
        if (more) fetch(kt + 32, nci0, nkh, nkw);   // loads in flight during WMMA below

        // ---- fragments from buf[cur], same contiguous k order for A and B ----
        bf16x16 af[2], bfm[4];
#pragma unroll
        for (int i = 0; i < 2; ++i) {
            // A 16x32: lanes 0-15 M=lane, K=0..15; lanes 16-31 K=16..31 (contiguous)
            const unsigned short* base = &lAs[cur * BUFE + (waveM * 32 + i * 16 + lm) * RS + lh * 16];
            FragU u;
            u.h[0] = *(const ushort8*)(base);
            u.h[1] = *(const ushort8*)(base + 8);
            af[i] = u.v;
        }
#pragma unroll
        for (int j = 0; j < 4; ++j) {
            // B 32x16: lanes 0-15 N=lane, K=0..15; lanes 16-31 K=16..31 (contiguous)
            const unsigned short* base = &lBs[cur * BUFE + (waveN * 64 + j * 16 + lm) * RS + lh * 16];
            FragU u;
            u.h[0] = *(const ushort8*)(base);
            u.h[1] = *(const ushort8*)(base + 8);
            bfm[j] = u.v;
        }

#pragma unroll
        for (int i = 0; i < 2; ++i)
#pragma unroll
            for (int j = 0; j < 4; ++j)
                acc[i][j] = __builtin_amdgcn_wmma_f32_16x16x32_bf16(
                    false, af[i], false, bfm[j], (short)0, acc[i][j], false, false);

        if (more) commit(cur ^ 1);             // waits on loads here, after WMMAs issued

        ci0 = nci0; kh = nkh; kw = nkw; cur ^= 1;
    }

    // ---- epilogue: bias or BN(+ReLU), scatter to NCHW ----
#pragma unroll
    for (int i = 0; i < 2; ++i) {
        const int coBase = m0 + waveM * 32 + i * 16 + lh * 8;  // C/D: VGPR r -> M=r(+8 hi half)
#pragma unroll
        for (int j = 0; j < 4; ++j) {
            const int n  = n0 + waveN * 64 + j * 16 + lm;
            const int bb = n >> 12;
            const int hw = n & 4095;
            float* outp = Out + (((size_t)bb * Co) << 12) + hw;
#pragma unroll
            for (int r = 0; r < 8; ++r) {
                const int co = coBase + r;
                float v = acc[i][j][r];
                if (EPI == 0) {
                    v += p0[co];
                } else {
                    float sc = p0[co] * rsqrtf(p3[co] + 1e-5f);
                    float sh = p1[co] - p2[co] * sc;
                    v = v * sc + sh;
                    if (EPI == 2) v = fmaxf(v, 0.0f);
                }
                outp[(size_t)co << 12] = v;
            }
        }
    }
}

// ---------------------------------------------------------------------------
// Deform: per-channel (dy,dx) offsets, clipped bilinear resample. Elementwise.
// ---------------------------------------------------------------------------
__global__ __launch_bounds__(256)
void deform_kernel(const float* __restrict__ x, const float* __restrict__ off,
                   float* __restrict__ out, int C, int Clog2) {
    const int idx = blockIdx.x * 256 + threadIdx.x;     // [0, 4*C*4096)
    const int w  = idx & 63;
    const int h  = (idx >> 6) & 63;
    const int bc = idx >> 12;
    const int c  = bc & (C - 1);
    const int b  = bc >> Clog2;

    const size_t base_x   = ((size_t)bc) << 12;
    const size_t base_off = ((size_t)(b * 2 * C + 2 * c)) << 12;
    const int hw = (h << 6) + w;

    float dy = off[base_off + hw];
    float dx = off[base_off + 4096 + hw];
    float py = fminf(fmaxf((float)h + dy, 0.0f), 63.0f);
    float px = fminf(fmaxf((float)w + dx, 0.0f), 63.0f);
    int y0 = (int)floorf(py), x0 = (int)floorf(px);
    int y1 = min(y0 + 1, 63),  x1 = min(x0 + 1, 63);
    float wy = py - (float)y0, wx = px - (float)x0;

    const float* xp = x + base_x;
    float v00 = xp[(y0 << 6) + x0];
    float v01 = xp[(y0 << 6) + x1];
    float v10 = xp[(y1 << 6) + x0];
    float v11 = xp[(y1 << 6) + x1];
    out[base_x + hw] = v00 * (1.f - wy) * (1.f - wx) + v01 * (1.f - wy) * wx +
                       v10 * wy * (1.f - wx)         + v11 * wy * wx;
}

// ---------------------------------------------------------------------------
extern "C" void kernel_launch(void* const* d_in, const int* in_sizes, int n_in,
                              void* d_out, int out_size, void* d_ws, size_t ws_size,
                              hipStream_t stream) {
    const float* x   = (const float*)d_in[0];
    const float* ow1 = (const float*)d_in[1];  const float* ob1 = (const float*)d_in[2];
    const float* ow2 = (const float*)d_in[3];  const float* ob2 = (const float*)d_in[4];
    const float* ow3 = (const float*)d_in[5];  const float* ob3 = (const float*)d_in[6];
    const float* w1  = (const float*)d_in[7];
    const float* w2  = (const float*)d_in[8];
    const float* w3  = (const float*)d_in[9];
    const float* g1 = (const float*)d_in[10], *b1 = (const float*)d_in[11],
               * m1 = (const float*)d_in[12], *v1 = (const float*)d_in[13];
    const float* g2 = (const float*)d_in[14], *b2 = (const float*)d_in[15],
               * m2 = (const float*)d_in[16], *v2 = (const float*)d_in[17];
    const float* g3 = (const float*)d_in[18], *b3 = (const float*)d_in[19],
               * m3 = (const float*)d_in[20], *v3 = (const float*)d_in[21];
    float* out = (float*)d_out;

    // workspace layout (stream-ordered reuse; activations peak 192 MiB)
    char* ws = (char*)d_ws;
    const size_t MB = 1024 * 1024;
    float* off1 = (float*)(ws);              // 128 MiB  [4,2048,64,64]
    float* h1   = (float*)(ws + 128 * MB);   //  64 MiB  [4,1024,64,64]
    float* h2   = (float*)(ws);              //  16 MiB  [4,256,64,64]  (off1 dead)
    float* off2 = (float*)(ws + 16 * MB);    //  32 MiB  [4,512,64,64]
    float* h3   = (float*)(ws + 48 * MB);    //  16 MiB
    float* h4   = (float*)(ws + 64 * MB);    //  16 MiB
    float* off3 = (float*)(ws + 80 * MB);    //  32 MiB
    float* h5   = (float*)(ws + 112 * MB);   //  16 MiB

    // repacked bf16 weights (~43 MiB) above the activation region
    unsigned short* q_ow1 = (unsigned short*)(ws + 192 * MB);   // 2048 x 9216
    unsigned short* q_w1  = q_ow1 + (size_t)2048 * 9216;        //  256 x 1024
    unsigned short* q_ow2 = q_w1  + (size_t)256 * 1024;         //  512 x 2304
    unsigned short* q_w2  = q_ow2 + (size_t)512 * 2304;         //  256 x 2304
    unsigned short* q_ow3 = q_w2  + (size_t)256 * 2304;         //  512 x 2304
    unsigned short* q_w3  = q_ow3 + (size_t)512 * 2304;         // 1024 x 256

    dim3 blk(256);
    const int NT = 128;                      // 16384 / 128 spatial tiles

    // 0) repack all weights to bf16, K-order (kh,kw,ci)
    repack_w<<<(2048 * 9216) / 256, blk, 0, stream>>>(ow1, q_ow1, 1024, 9);
    repack_w<<<(256 * 1024)  / 256, blk, 0, stream>>>(w1,  q_w1,  1024, 1);
    repack_w<<<(512 * 2304)  / 256, blk, 0, stream>>>(ow2, q_ow2, 256,  9);
    repack_w<<<(256 * 2304)  / 256, blk, 0, stream>>>(w2,  q_w2,  256,  9);
    repack_w<<<(512 * 2304)  / 256, blk, 0, stream>>>(ow3, q_ow3, 256,  9);
    repack_w<<<(1024 * 256)  / 256, blk, 0, stream>>>(w3,  q_w3,  256,  1);

    // 1) off1 = conv3x3(x; ow1) + ob1           (1024 -> 2048, K=9216)
    conv_gemm_wmma<9, 0><<<dim3(NT, 2048 / 128), blk, 0, stream>>>(
        q_ow1, x, off1, 1024, 9216, 2048, ob1, nullptr, nullptr, nullptr);
    // 2) h1 = deform(x, off1)
    deform_kernel<<<(4 * 1024 * 4096) / 256, blk, 0, stream>>>(x, off1, h1, 1024, 10);
    // 3) h2 = BN1(conv1x1(h1; w1))              (1024 -> 256, K=1024)
    conv_gemm_wmma<1, 1><<<dim3(NT, 256 / 128), blk, 0, stream>>>(
        q_w1, h1, h2, 1024, 1024, 256, g1, b1, m1, v1);
    // 4) off2 = conv3x3(h2; ow2) + ob2          (256 -> 512, K=2304)
    conv_gemm_wmma<9, 0><<<dim3(NT, 512 / 128), blk, 0, stream>>>(
        q_ow2, h2, off2, 256, 2304, 512, ob2, nullptr, nullptr, nullptr);
    // 5) h3 = deform(h2, off2)
    deform_kernel<<<(4 * 256 * 4096) / 256, blk, 0, stream>>>(h2, off2, h3, 256, 8);
    // 6) h4 = BN2(conv3x3(h3; w2))              (256 -> 256, K=2304)
    conv_gemm_wmma<9, 1><<<dim3(NT, 256 / 128), blk, 0, stream>>>(
        q_w2, h3, h4, 256, 2304, 256, g2, b2, m2, v2);
    // 7) off3 = conv3x3(h4; ow3) + ob3          (256 -> 512, K=2304)
    conv_gemm_wmma<9, 0><<<dim3(NT, 512 / 128), blk, 0, stream>>>(
        q_ow3, h4, off3, 256, 2304, 512, ob3, nullptr, nullptr, nullptr);
    // 8) h5 = deform(h4, off3)
    deform_kernel<<<(4 * 256 * 4096) / 256, blk, 0, stream>>>(h4, off3, h5, 256, 8);
    // 9) out = relu(BN3(conv1x1(h5; w3)))       (256 -> 1024, K=256)
    conv_gemm_wmma<1, 2><<<dim3(NT, 1024 / 128), blk, 0, stream>>>(
        q_w3, h5, out, 256, 256, 1024, g3, b3, m3, v3);
}
